// SpatialS4DEEG_15470472200788
// MI455X (gfx1250) — compile-verified, matrix operand order checked by reading, not run on hardware
//
#include <hip/hip_runtime.h>
#include <hip/hip_bf16.h>

typedef __attribute__((ext_vector_type(16))) _Float16 v16h;
typedef __attribute__((ext_vector_type(8)))  float    v8f;
typedef __attribute__((ext_vector_type(4)))  int      v4i;

#define DEV __device__ __forceinline__

#if defined(__HIP_DEVICE_COMPILE__) && __has_builtin(__builtin_amdgcn_global_load_async_to_lds_b128)
#define HAVE_ASYNC_LDS 1
typedef __attribute__((address_space(1))) v4i gv4i_t;  // global-AS 4-dword payload
typedef __attribute__((address_space(3))) v4i lv4i_t;  // LDS-AS 4-dword payload
#endif

// ---------------- WMMA fragment helpers (CDNA5 wave32, 16x16x32 f16) -------
// A-frag: lane holds row M=lane&15; element e -> K = e + 8*(lane>=16) + 8*(e>=8)
//         (two contiguous runs of 8 halves -> mergeable into 2x b128 loads)
// B-frag: lane holds col N=lane&15; same K mapping.
// C/D:    vgpr r -> row M = r + 8*(lane>=16), col N = lane&15.

template <typename G>
DEV v16h load_frag(int lane, G gather) {
  v16h r;
  int hi = lane >> 4, mn = lane & 15;
#pragma unroll
  for (int e = 0; e < 16; ++e) {
    int k = e + hi * 8 + (e >= 8 ? 8 : 0);
    r[e] = gather(mn, k);
  }
  return r;
}

template <typename S>
DEV void store_frag(int lane, v8f c, S st) {
  int n = lane & 15, hi = lane >> 4;
#pragma unroll
  for (int r = 0; r < 8; ++r) st(r + hi * 8, n, c[r]);
}

DEV v8f wmma32(v16h a, v16h b, v8f c) {
  return __builtin_amdgcn_wmma_f32_16x16x32_f16(false, a, false, b, (short)0, c,
                                                false, false);
}

DEV float sigf(float x) { return 1.f / (1.f + __expf(-x)); }
DEV float geluf(float x) {
  float t = 0.7978845608f * (x + 0.044715f * x * x * x);
  float e = __expf(2.f * t);
  return 0.5f * x * (1.f + (e - 1.f) / (e + 1.f));
}

// ---------------- 1. channel attention ------------------------------------
__global__ void k_chanattn(const float* __restrict__ x, const float* __restrict__ w1,
                           const float* __restrict__ b1, const float* __restrict__ w2,
                           const float* __restrict__ b2, float* __restrict__ cw, int T) {
  __shared__ float sm[128], sh[64];
  int b = blockIdx.x, tid = threadIdx.x;  // 128 threads
  const float* xr = x + (size_t)b * 129 * T + (size_t)tid * T;
  float s = 0.f;
  for (int t = 0; t < T; ++t) s += xr[t];
  sm[tid] = s / (float)T;
  __syncthreads();
  if (tid < 64) {
    float a = b1[tid];
    for (int c = 0; c < 128; ++c) a += w1[tid * 128 + c] * sm[c];
    sh[tid] = fmaxf(a, 0.f);
  }
  __syncthreads();
  float a = b2[tid];
  for (int j = 0; j < 64; ++j) a += w2[tid * 64 + j] * sh[j];
  cw[(size_t)b * 128 + tid] = sigf(a);
}

// ---------------- 2. spatial projection + BN + ELU (WMMA) ------------------
// xs[b,f,t] = elu(bn(sum_c spatial_w[f,c] * eeg[b,c,t] * cw[b,c]))
// Activation panel staged TRANSPOSED (t-major, row stride 130 halves: odd
// dword stride -> conflict-free banks) so B-fragments are stride-1 in K.
__global__ void k_spatial(const float* __restrict__ x, const float* __restrict__ cw,
                          const float* __restrict__ sw, const float* __restrict__ bg,
                          const float* __restrict__ bb, const float* __restrict__ bm,
                          const float* __restrict__ bv, float* __restrict__ xs, int T) {
  extern __shared__ _Float16 lds_sp[];
  _Float16* XhT = lds_sp;                // 208 x 130 (N x K, padded)
  _Float16* Wh = lds_sp + 208 * 130;     // 48 x 128  (M x K)
  int b = blockIdx.x, tid = threadIdx.x;  // 256 threads
  for (int i = tid; i < 48 * 128; i += 256) {
    int f = i >> 7, c = i & 127;
    Wh[i] = (f < 36) ? (_Float16)sw[f * 128 + c] : (_Float16)0.f;
  }
  for (int i = tid; i < 128 * 208; i += 256) {
    int c = i / 208, t = i - c * 208;
    float v = (t < T && c < 128)
                  ? x[(size_t)b * 129 * T + (size_t)c * T + t] * cw[(size_t)b * 128 + c]
                  : 0.f;
    XhT[t * 130 + c] = (_Float16)v;
  }
  __syncthreads();
  int wave = tid >> 5, lane = tid & 31;
  for (int tile = wave; tile < 3 * 13; tile += 8) {
    int m0 = (tile / 13) * 16, n0 = (tile % 13) * 16;
    v8f acc = {};
    for (int k0 = 0; k0 < 128; k0 += 32) {
      v16h a = load_frag(lane, [&](int m, int k) { return Wh[(m0 + m) * 128 + k0 + k]; });
      v16h bf = load_frag(lane, [&](int n, int k) { return XhT[(n0 + n) * 130 + k0 + k]; });
      acc = wmma32(a, bf, acc);
    }
    store_frag(lane, acc, [&](int r, int n, float v) {
      int f = m0 + r, t = n0 + n;
      if (f < 36 && t < T) {
        float z = (v - bm[f]) * (bg[f] * rsqrtf(bv[f] + 1e-5f)) + bb[f];
        xs[(size_t)b * 36 * T + f * T + t] = (z > 0.f) ? z : (__expf(z) - 1.f);
      }
    });
  }
}

// ---------------- 3. generic conv1d as implicit GEMM (WMMA) ----------------
// One block per sample: full zero-padded time rows staged once, all (oc,t)
// tiles computed from it. B-fragment gathers are branchless and stride-1.
__global__ void k_conv1d(const float* __restrict__ in, size_t inBs, int inCs, int inTs,
                         const float* __restrict__ w, const float* __restrict__ bias,
                         float* __restrict__ out, size_t outBs, int outCs, int outTs,
                         int ocoff, int IC, int OC, int KS, int padL, int T, int relu) {
  extern __shared__ _Float16 Xs[];  // IC x LP  (LP = T + KP + 32)
  int b = blockIdx.x, tid = threadIdx.x;
  int KP = ((KS + 31) / 32) * 32;
  int LP = T + KP + 32;
  for (int i = tid; i < IC * LP; i += blockDim.x) {
    int ic = i / LP, j = i - ic * LP;
    int ti = j - padL;
    Xs[i] = (ti >= 0 && ti < T)
                ? (_Float16)in[b * inBs + (size_t)ic * inCs + (size_t)ti * inTs]
                : (_Float16)0.f;
  }
  __syncthreads();
  int wave = tid >> 5, lane = tid & 31, nwaves = blockDim.x >> 5;
  int nT = (T + 15) / 16, nOcT = (OC + 15) / 16, ntiles = nT * nOcT;
  for (int tile = wave; tile < ntiles; tile += nwaves) {
    int ot = tile / nT, tt = tile - ot * nT;
    int oc0 = ot * 16, t0 = tt * 16;
    int ocA = oc0 + (lane & 15);
    const float* wrow = w + (size_t)ocA * IC * KS;
    v8f acc = {};
    for (int ic = 0; ic < IC; ++ic) {
      const float* wr = wrow + (size_t)ic * KS;
      const _Float16* xr = Xs + ic * LP + t0;
      for (int k0 = 0; k0 < KP; k0 += 32) {
        v16h a = load_frag(lane, [&](int m, int k) {
          int kk = k0 + k;
          return (ocA < OC && kk < KS) ? (_Float16)wr[kk] : (_Float16)0.f;
        });
        v16h bf = load_frag(lane, [&](int n, int k) { return xr[n + k0 + k]; });
        acc = wmma32(a, bf, acc);
      }
    }
    store_frag(lane, acc, [&](int r, int n, float v) {
      int oc = oc0 + r, t = t0 + n;
      if (oc < OC && t < T) {
        float z = v + bias[oc];
        out[b * outBs + (size_t)(ocoff + oc) * outCs + (size_t)t * outTs] =
            relu ? fmaxf(z, 0.f) : z;
      }
    });
  }
}

// ---------------- 4. fusion matmul + double LayerNorm -> v1 (B,36,T) -------
__global__ void k_fusion_ln(const float* __restrict__ xs, const float* __restrict__ xf,
                            const float* __restrict__ fw, const float* __restrict__ fb,
                            const float* __restrict__ g1, const float* __restrict__ b1,
                            const float* __restrict__ g2, const float* __restrict__ b2,
                            float* __restrict__ v, int T) {
  extern __shared__ float sm4[];  // 36*T (xf slice)
  int b = blockIdx.x, tid = threadIdx.x;
#if defined(HAVE_ASYNC_LDS)
  {  // async global->LDS copy of the verbatim f32 panel (7200 floats)
    const float* src = xf + (size_t)b * 36 * T;
    for (int i = tid; i < (36 * T) / 4; i += blockDim.x)
      __builtin_amdgcn_global_load_async_to_lds_b128((gv4i_t*)(src + 4 * i),
                                                     (lv4i_t*)(sm4 + 4 * i), 0, 0);
    asm volatile("s_wait_asynccnt 0x0" ::: "memory");
  }
#else
  for (int i = tid; i < 36 * T; i += blockDim.x) sm4[i] = xf[(size_t)b * 36 * T + i];
#endif
  __syncthreads();
  for (int t = tid; t < T; t += blockDim.x) {
    float col[36], u[36];
    for (int f = 0; f < 36; ++f) {
      float s = fb[f];
      for (int c = 0; c < 36; ++c) s += fw[f * 36 + c] * sm4[c * T + t];
      col[f] = s + xs[(size_t)b * 36 * T + f * T + t];
    }
    float m = 0.f;
    for (int f = 0; f < 36; ++f) m += col[f];
    m /= 36.f;
    float var = 0.f;
    for (int f = 0; f < 36; ++f) { float d = col[f] - m; var += d * d; }
    float inv = rsqrtf(var / 36.f + 1e-5f);
    for (int f = 0; f < 36; ++f) u[f] = (col[f] - m) * inv * g1[f] + b1[f];
    m = 0.f;
    for (int f = 0; f < 36; ++f) m += u[f];
    m /= 36.f;
    var = 0.f;
    for (int f = 0; f < 36; ++f) { float d = u[f] - m; var += d * d; }
    inv = rsqrtf(var / 36.f + 1e-5f);
    for (int f = 0; f < 36; ++f)
      v[(size_t)b * 36 * T + f * T + t] = (u[f] - m) * inv * g2[f] + b2[f];
  }
}

// ---------------- 5. S4D kernel generation (transcendental VALU) -----------
__global__ void k_s4dkern(const float* __restrict__ log_dt, const float* __restrict__ lAre,
                          const float* __restrict__ Aim, const float* __restrict__ Cre,
                          const float* __restrict__ Cim, int N, int T,
                          float* __restrict__ K) {
  int d = blockIdx.x, t = threadIdx.x;
  if (t >= T) return;
  float dt = __expf(log_dt[d]);
  float acc = 0.f;
  for (int n = 0; n < N; ++n) {
    float Are = -__expf(lAre[d * N + n]), Ai = Aim[d * N + n];
    float dAre = dt * Are, dAim = dt * Ai;
    float er = __expf(dAre), sn, cs;
    __sincosf(dAim, &sn, &cs);
    float Ere = er * cs - 1.f, Eim = er * sn;          // exp(dtA)-1
    float den = Are * Are + Ai * Ai;
    float Fr = (Ere * Are + Eim * Ai) / den;           // (exp(dtA)-1)/A
    float Fi = (Eim * Are - Ere * Ai) / den;
    float cr = Cre[d * N + n], ci = Cim[d * N + n];
    float CR = cr * Fr - ci * Fi, CI = cr * Fi + ci * Fr;
    float vr = __expf(dAre * t), s2, c2;
    __sincosf(dAim * t, &s2, &c2);
    acc += CR * (vr * c2) - CI * (vr * s2);            // Re(C * exp(dtA*t))
  }
  K[d * T + t] = 2.f * acc;
}

// ---------------- 6. causal conv as Toeplitz GEMM (WMMA) -------------------
// fwd: y[b,t] = sum_{t'<=t} K[t-t']*v[b,t'] + Dd*v[b,t]
// bwd: y[b,t] = sum_{t'>=t} K[t'-t]*v[b,t'] + Dd*v[b,t]
// A panel (16 batch rows x 224) staged once per block; Toeplitz kernel row
// staged into a 448-wide zero-padded array so B-frags are branchless stride-1.
// Requires T <= 200 (OFF=224 padding).
__global__ void k_ssm(const float* __restrict__ v, const float* __restrict__ K,
                      const float* __restrict__ Dp, float* __restrict__ y, int D, int T,
                      int backward, int ocoff, int outCtot) {
  __shared__ _Float16 Va[16 * 224];
  __shared__ _Float16 Kpad[448];
  const int KP = 224, OFF = 224;
  int d = blockIdx.x, m0 = blockIdx.y * 16;
  int lane = threadIdx.x;
  __builtin_prefetch(&v[((size_t)(m0 + (lane & 15)) * D + d) * T], 0, 0);
  for (int i = lane; i < 448; i += 32) {
    int dd = backward ? (i - OFF) : (OFF - i);  // Kpad[tp - t + OFF] = K[dd]
    Kpad[i] = (dd >= 0 && dd < T) ? (_Float16)K[d * T + dd] : (_Float16)0.f;
  }
  for (int i = lane; i < 16 * KP; i += 32) {
    int m = i / KP, tp = i - m * KP;
    Va[i] = (tp < T) ? (_Float16)v[((size_t)(m0 + m) * D + d) * T + tp] : (_Float16)0.f;
  }
  __syncthreads();
  float Dd = Dp[d];
  int nT = (T + 15) / 16;
  for (int nt = 0; nt < nT; ++nt) {
    int n0 = nt * 16;
    int base = OFF - (n0 + (lane & 15));
    v8f acc = {};
    for (int k0 = 0; k0 < KP; k0 += 32) {
      v16h a = load_frag(lane, [&](int m, int k) { return Va[m * KP + k0 + k]; });
      v16h bf = load_frag(lane, [&](int n, int k) { return Kpad[base + k0 + k]; });
      acc = wmma32(a, bf, acc);
    }
    store_frag(lane, acc, [&](int r, int n, float val) {
      int bb = m0 + r, t = n0 + n;
      if (t < T)
        y[((size_t)bb * outCtot + ocoff + d) * T + t] =
            val + Dd * v[((size_t)bb * D + d) * T + t];
    });
  }
}

// ---------------- 7. gelu(y)@W^T + bias (+resid) -> u (B,T,72)  (WMMA) -----
__global__ void k_gelu_proj(const float* __restrict__ y, const float* __restrict__ W,
                            const float* __restrict__ bias, const float* __restrict__ resid,
                            float* __restrict__ u, int Dch, int T) {
  int m0 = blockIdx.x * 16, n0 = blockIdx.y * 16, lane = threadIdx.x;
  int rowA = m0 + (lane & 15);
  int bA = rowA / T, tA = rowA - bA * T;
  const float* yr = y + (size_t)bA * Dch * T + tA;  // element k at yr[k*T]
  int colB = n0 + (lane & 15);
  const float* wr = W + (size_t)(colB < 72 ? colB : 0) * Dch;
  v8f acc = {};
  for (int k0 = 0; k0 < 96; k0 += 32) {
    v16h a = load_frag(lane, [&](int m, int k) {
      int kk = k0 + k;
      return (kk < Dch) ? (_Float16)geluf(yr[(size_t)kk * T]) : (_Float16)0.f;
    });
    v16h bf = load_frag(lane, [&](int n, int k) {
      int kk = k0 + k;
      return (kk < Dch && colB < 72) ? (_Float16)wr[kk] : (_Float16)0.f;
    });
    acc = wmma32(a, bf, acc);
  }
  store_frag(lane, acc, [&](int r, int n, float val) {
    int row = m0 + r, col = n0 + n;
    if (col < 72) {
      float z = val + bias[col];
      if (resid) z += resid[(size_t)row * 72 + col];
      u[(size_t)row * 72 + col] = z;
    }
  });
}

// ---------------- 8. LayerNorm over channels + transpose -> (B,D,T) --------
__global__ void k_ln_t(const float* __restrict__ u, const float* __restrict__ g,
                       const float* __restrict__ be, float* __restrict__ v, int D, int T,
                       int BT) {
  int idx = blockIdx.x * blockDim.x + threadIdx.x;
  if (idx >= BT) return;
  int b = idx / T, t = idx - b * T;
  const float* ur = u + (size_t)idx * D;
  float m = 0.f;
  for (int c = 0; c < D; ++c) m += ur[c];
  m /= (float)D;
  float var = 0.f;
  for (int c = 0; c < D; ++c) { float d = ur[c] - m; var += d * d; }
  float inv = rsqrtf(var / (float)D + 1e-5f);
  float* vb = v + (size_t)b * D * T + t;
  for (int c = 0; c < D; ++c) vb[(size_t)c * T] = (ur[c] - m) * inv * g[c] + be[c];
}

// ---------------- 9. attention softmax + weighted pooling ------------------
__global__ void k_attnpool(const float* __restrict__ u2, const float* __restrict__ h,
                           const float* __restrict__ w2, const float* __restrict__ b2s,
                           const float* __restrict__ tw, float* __restrict__ pooled, int T) {
  __shared__ float sl[256], red[256];
  int b = blockIdx.x, tid = threadIdx.x;  // 256 threads
  float lv = -3.0e38f;
  if (tid < T) {
    float a = b2s[0];
    for (int c = 0; c < 36; ++c) a += h[((size_t)b * 36 + c) * T + tid] * w2[c];
    lv = sigf(a) * sigf(tw[tid]);
  }
  sl[tid] = lv;
  red[tid] = lv;
  __syncthreads();
  for (int s = 128; s > 0; s >>= 1) {
    if (tid < s) red[tid] = fmaxf(red[tid], red[tid + s]);
    __syncthreads();
  }
  float mx = red[0];
  __syncthreads();
  float ev = (tid < T) ? __expf(sl[tid] - mx) : 0.f;
  red[tid] = ev;
  __syncthreads();
  for (int s = 128; s > 0; s >>= 1) {
    if (tid < s) red[tid] += red[tid + s];
    __syncthreads();
  }
  float denom = red[0];
  __syncthreads();
  sl[tid] = ev / denom;
  __syncthreads();
  int G = T / 8;
  for (int i = tid; i < 576; i += blockDim.x) {
    int c = i >> 3, gp = i & 7;
    float s = 0.f;
    for (int j = 0; j < G; ++j) {
      int t = gp * G + j;
      s += u2[((size_t)b * T + t) * 72 + c] * sl[t];
    }
    pooled[(size_t)b * 576 + i] = s / (float)G;
  }
}

// ---------------- 10. FC layer (WMMA), requires M%16==N%16==Kd%32==0 -------
__global__ void k_fc(const float* __restrict__ A, const float* __restrict__ W,
                     const float* __restrict__ bias, float* __restrict__ out, int M, int N,
                     int Kd, int relu) {
  int m0 = blockIdx.x * 16, n0 = blockIdx.y * 16, lane = threadIdx.x;
  const float* Ar = A + (size_t)(m0 + (lane & 15)) * Kd;
  const float* Wr = W + (size_t)(n0 + (lane & 15)) * Kd;
  v8f acc = {};
  for (int k0 = 0; k0 < Kd; k0 += 32) {
    v16h a = load_frag(lane, [&](int m, int k) { return (_Float16)Ar[k0 + k]; });
    v16h bf = load_frag(lane, [&](int n, int k) { return (_Float16)Wr[k0 + k]; });
    acc = wmma32(a, bf, acc);
  }
  store_frag(lane, acc, [&](int r, int n, float v) {
    float z = v + bias[n0 + n];
    out[(size_t)(m0 + r) * N + n0 + n] = relu ? fmaxf(z, 0.f) : z;
  });
}

// ---------------- 11. LayerNorm(128) + relu --------------------------------
__global__ void k_ln_relu(const float* __restrict__ in, const float* __restrict__ g,
                          const float* __restrict__ be, float* __restrict__ out, int D) {
  __shared__ float red[128];
  int b = blockIdx.x, tid = threadIdx.x;  // 128 threads
  float x = in[(size_t)b * D + tid];
  red[tid] = x;
  __syncthreads();
  for (int s = 64; s > 0; s >>= 1) {
    if (tid < s) red[tid] += red[tid + s];
    __syncthreads();
  }
  float m = red[0] / (float)D;
  __syncthreads();
  float d = x - m;
  red[tid] = d * d;
  __syncthreads();
  for (int s = 64; s > 0; s >>= 1) {
    if (tid < s) red[tid] += red[tid + s];
    __syncthreads();
  }
  float inv = rsqrtf(red[0] / (float)D + 1e-5f);
  out[(size_t)b * D + tid] = fmaxf(d * inv * g[tid] + be[tid], 0.f);
}

// ---------------- 12. final head (N=1) -------------------------------------
__global__ void k_head(const float* __restrict__ h3, const float* __restrict__ w4,
                       const float* __restrict__ b4, float* __restrict__ out, int B) {
  int b = blockIdx.x * blockDim.x + threadIdx.x;
  if (b >= B) return;
  float s = b4[0];
  for (int j = 0; j < 32; ++j) s += h3[(size_t)b * 32 + j] * w4[j];
  out[b] = s;
}

// ===========================================================================
extern "C" void kernel_launch(void* const* d_in, const int* in_sizes, int n_in,
                              void* d_out, int out_size, void* d_ws, size_t ws_size,
                              hipStream_t stream) {
  const float* x = (const float*)d_in[0];
  const float* ca_w1 = (const float*)d_in[1];
  const float* ca_b1 = (const float*)d_in[2];
  const float* ca_w2 = (const float*)d_in[3];
  const float* ca_b2 = (const float*)d_in[4];
  const float* spatial_w = (const float*)d_in[5];
  const float* bn_g = (const float*)d_in[6];
  const float* bn_b = (const float*)d_in[7];
  const float* bn_m = (const float*)d_in[8];
  const float* bn_v = (const float*)d_in[9];
  const float* fb_w[5] = {(const float*)d_in[10], (const float*)d_in[12],
                          (const float*)d_in[14], (const float*)d_in[16],
                          (const float*)d_in[18]};
  const float* fb_b[5] = {(const float*)d_in[11], (const float*)d_in[13],
                          (const float*)d_in[15], (const float*)d_in[17],
                          (const float*)d_in[19]};
  const float* fusion_w = (const float*)d_in[20];
  const float* fusion_b = (const float*)d_in[21];
  const float* fd_g = (const float*)d_in[22];
  const float* fd_b = (const float*)d_in[23];
  const float* s1_g = (const float*)d_in[24];
  const float* s1_b = (const float*)d_in[25];
  const float* s1_logdt = (const float*)d_in[26];
  const float* s1_lAre = (const float*)d_in[27];
  const float* s1_Aim = (const float*)d_in[28];
  const float* s1_ow = (const float*)d_in[29];
  const float* s1_ob = (const float*)d_in[30];
  const float* s2_g = (const float*)d_in[31];
  const float* s2_b = (const float*)d_in[32];
  const float* s2_logdt = (const float*)d_in[33];
  const float* s2_lAre = (const float*)d_in[34];
  const float* s2_Aim = (const float*)d_in[35];
  const float* s2_ow = (const float*)d_in[36];
  const float* s2_ob = (const float*)d_in[37];
  const float* s1_Cref = (const float*)d_in[38];
  const float* s1_Cimf = (const float*)d_in[39];
  const float* s1_Creb = (const float*)d_in[40];
  const float* s1_Cimb = (const float*)d_in[41];
  const float* s1_Df = (const float*)d_in[42];
  const float* s1_Db = (const float*)d_in[43];
  const float* s2_Cre = (const float*)d_in[44];
  const float* s2_Cim = (const float*)d_in[45];
  const float* s2_D = (const float*)d_in[46];
  const float* ra_w1 = (const float*)d_in[47];
  const float* ra_b1 = (const float*)d_in[48];
  const float* ra_w2 = (const float*)d_in[49];
  const float* ra_b2 = (const float*)d_in[50];
  const float* tw = (const float*)d_in[51];
  const float* p_w1 = (const float*)d_in[52];
  const float* p_b1 = (const float*)d_in[53];
  const float* p_ln_g = (const float*)d_in[54];
  const float* p_ln_b = (const float*)d_in[55];
  const float* p_w2 = (const float*)d_in[56];
  const float* p_b2 = (const float*)d_in[57];
  const float* p_w3 = (const float*)d_in[58];
  const float* p_b3 = (const float*)d_in[59];
  const float* p_w4 = (const float*)d_in[60];
  const float* p_b4 = (const float*)d_in[61];

  const int T = 200;
  int B = in_sizes[0] / (129 * T);
  int BT = B * T;

  // workspace bump allocator
  char* wsp = (char*)d_ws;
  auto alloc = [&](size_t nfloats) {
    float* p = (float*)wsp;
    wsp += ((nfloats * sizeof(float) + 255) & ~(size_t)255);
    return p;
  };
  float* cw = alloc((size_t)B * 128);
  float* xs = alloc((size_t)B * 36 * T);
  float* xf = alloc((size_t)B * 36 * T);
  float* v1 = alloc((size_t)B * 36 * T);
  float* Kf = alloc((size_t)36 * T);
  float* Kb = alloc((size_t)36 * T);
  float* K2 = alloc((size_t)72 * T);
  float* ybuf = alloc((size_t)B * 72 * T);  // reused: y1 then y2
  float* u1 = alloc((size_t)B * T * 72);
  float* v2 = alloc((size_t)B * 72 * T);
  float* u2 = alloc((size_t)B * T * 72);
  float* hra = alloc((size_t)B * 36 * T);
  float* pooled = alloc((size_t)B * 576);
  float* fc1 = alloc((size_t)B * 128);
  float* fc1n = alloc((size_t)B * 128);
  float* fc2 = alloc((size_t)B * 64);
  float* fc3 = alloc((size_t)B * 32);

  // 1. channel attention
  k_chanattn<<<B, 128, 0, stream>>>(x, ca_w1, ca_b1, ca_w2, ca_b2, cw, T);

  // 2. spatial projection + BN + ELU
  size_t lds_sp = (size_t)(208 * 130 + 48 * 128) * sizeof(_Float16);
  k_spatial<<<B, 256, lds_sp, stream>>>(x, cw, spatial_w, bn_g, bn_b, bn_m, bn_v, xs, T);

  // 3. filterbank convs -> xf (channel offsets 0,8,15,22,29)
  const int boc[5] = {8, 7, 7, 7, 7}, bks[5] = {200, 50, 25, 10, 5};
  int ocoff = 0;
  for (int i = 0; i < 5; ++i) {
    int KS = bks[i];
    int KP = ((KS + 31) / 32) * 32;
    size_t lds = (size_t)36 * (T + KP + 32) * sizeof(_Float16);
    k_conv1d<<<B, 256, lds, stream>>>(xs, (size_t)36 * T, T, 1, fb_w[i], fb_b[i], xf,
                                      (size_t)36 * T, T, 1, ocoff, 36, boc[i], KS,
                                      (KS - 1) / 2, T, 0);
    ocoff += boc[i];
  }

  // 4. fusion + double LayerNorm -> v1 (B,36,T)
  k_fusion_ln<<<B, 256, (size_t)36 * T * sizeof(float), stream>>>(
      xs, xf, fusion_w, fusion_b, fd_g, fd_b, s1_g, s1_b, v1, T);

  // 5. S4D kernels
  k_s4dkern<<<36, 256, 0, stream>>>(s1_logdt, s1_lAre, s1_Aim, s1_Cref, s1_Cimf, 32, T, Kf);
  k_s4dkern<<<36, 256, 0, stream>>>(s1_logdt, s1_lAre, s1_Aim, s1_Creb, s1_Cimb, 32, T, Kb);
  k_s4dkern<<<72, 256, 0, stream>>>(s2_logdt, s2_lAre, s2_Aim, s2_Cre, s2_Cim, 32, T, K2);

  // 6. S4D layer 1: bidirectional Toeplitz GEMMs -> ybuf (B,72,T)
  dim3 gssm1(36, B / 16);
  k_ssm<<<gssm1, 32, 0, stream>>>(v1, Kf, s1_Df, ybuf, 36, T, 0, 0, 72);
  k_ssm<<<gssm1, 32, 0, stream>>>(v1, Kb, s1_Db, ybuf, 36, T, 1, 36, 72);

  // 7. u1 = gelu(y1) @ s1_out_w^T + b
  dim3 gproj(BT / 16, 5);
  k_gelu_proj<<<gproj, 32, 0, stream>>>(ybuf, s1_ow, s1_ob, nullptr, u1, 72, T);

  // 8. v2 = LN(u1) transposed -> (B,72,T)
  k_ln_t<<<(BT + 255) / 256, 256, 0, stream>>>(u1, s2_g, s2_b, v2, 72, T, BT);

  // 9. S4D layer 2 + residual projection -> u2
  dim3 gssm2(72, B / 16);
  k_ssm<<<gssm2, 32, 0, stream>>>(v2, K2, s2_D, ybuf, 72, T, 0, 0, 72);
  k_gelu_proj<<<gproj, 32, 0, stream>>>(ybuf, s2_ow, s2_ob, u1, u2, 72, T);

  // 10. residual attention conv (reads u2 in (B,T,72) layout via strides)
  {
    int KS = 25, KP = 32;
    size_t lds_ra = (size_t)72 * (T + KP + 32) * sizeof(_Float16);
    k_conv1d<<<B, 256, lds_ra, stream>>>(u2, (size_t)T * 72, 1, 72, ra_w1, ra_b1, hra,
                                         (size_t)36 * T, T, 1, 0, 72, 36, KS,
                                         (KS - 1) / 2, T, 1);
  }

  // 11. softmax attention + pooled (B,576)
  k_attnpool<<<B, 256, 0, stream>>>(u2, hra, ra_w2, ra_b2, tw, pooled, T);

  // 12. MLP head
  k_fc<<<dim3(B / 16, 8), 32, 0, stream>>>(pooled, p_w1, p_b1, fc1, B, 128, 576, 0);
  k_ln_relu<<<B, 128, 0, stream>>>(fc1, p_ln_g, p_ln_b, fc1n, 128);
  k_fc<<<dim3(B / 16, 4), 32, 0, stream>>>(fc1n, p_w2, p_b2, fc2, B, 64, 128, 1);
  k_fc<<<dim3(B / 16, 2), 32, 0, stream>>>(fc2, p_w3, p_b3, fc3, B, 32, 64, 1);
  k_head<<<(B + 255) / 256, 256, 0, stream>>>(fc3, p_w4, p_b4, (float*)d_out, B);
}
